// MechGNN_31851477467891
// MI455X (gfx1250) — compile-verified, matrix operand order checked by reading, not run on hardware
//
#include <hip/hip_runtime.h>
#include <hip/hip_bf16.h>
#include <math.h>

// ---------------------------------------------------------------------------
// CDNA5 (gfx1250) AttentiveFP forward. All dense layers run through
// v_wmma_f32_16x16x32_f16 (f32 activations converted to f16 in LDS).
// 64x64 block tile, 8 waves, 16x32 per wave (2 accumulators, shared A frag).
// LDS fragments are fetched as ds_load_b128 pairs (B stored transposed).
// ---------------------------------------------------------------------------

typedef __attribute__((ext_vector_type(16))) _Float16 v16h;
typedef __attribute__((ext_vector_type(8)))  _Float16 v8h;
typedef __attribute__((ext_vector_type(4)))  _Float16 v4h;
typedef __attribute__((ext_vector_type(8)))  float    v8f;

constexpr int HD  = 256;   // hidden dim
constexpr int GH3 = 768;   // 3*H for GRU gates
constexpr int TBM = 64;    // block tile M (4 waves)
constexpr int TBN = 64;    // block tile N (2 waves x 2 subtiles)
constexpr int TKT = 32;    // K step (WMMA K)
constexpr int LDA = 40;    // LDS stride (halves): 80B pitch -> 16B aligned
constexpr int LDB = 40;

__device__ __forceinline__ float leaky01(float v){ return v > 0.f ? v : 0.01f * v; }
__device__ __forceinline__ float elu1(float v)   { return v > 0.f ? v : (__expf(v) - 1.f); }
__device__ __forceinline__ float sigmf(float v)  { return 1.f / (1.f + __expf(-v)); }

__device__ __forceinline__ void atomicMaxF(float* addr, float v) {
  unsigned int* ua = (unsigned int*)addr;
  unsigned int old = __float_as_uint(*addr);
  while (__uint_as_float(old) < v) {
    unsigned int assumed = old;
    old = atomicCAS(ua, assumed, __float_as_uint(v));
    if (old == assumed) break;
  }
}

// Load one lane's v16h fragment: two contiguous 8-half runs (2x ds_load_b128).
// base points at the lane's LDS row (16B aligned); kb in {0,8} halves.
__device__ __forceinline__ v16h load_frag(const _Float16* base, int kb) {
  v16h f;
  ((v8h*)&f)[0] = *(const v8h*)(base + kb);        // K = kb .. kb+7
  ((v8h*)&f)[1] = *(const v8h*)(base + kb + 16);   // K = kb+16 .. kb+23
  return f;
}

__device__ __forceinline__ float apply_act(float v, int ACT) {
  if (ACT == 1) return fmaxf(v, 0.f);
  if (ACT == 2) return leaky01(v);
  if (ACT == 3) return elu1(v);
  return v;
}

// ---------------------------------------------------------------------------
// C[M,Nc] = act(A[M,K] @ W[K,Nc] + bias). Nc must be a multiple of 64.
// K arbitrary (zero padded into LDS; vector path when K%4==0).
// ---------------------------------------------------------------------------
template<int ACT>
__global__ __launch_bounds__(256)
void gemm_wmma(const float* __restrict__ A, const float* __restrict__ W,
               const float* __restrict__ bias, float* __restrict__ C,
               int M, int K, int Nc)
{
  __shared__ __align__(16) _Float16 sA [TBM * LDA];   // [m][k]
  __shared__ __align__(16) _Float16 sBt[TBN * LDB];   // [n][k] (transposed)
  const int tid  = threadIdx.x;
  const int lane = tid & 31;
  const int wave = tid >> 5;
  const int wm   = wave >> 1;          // 0..3
  const int wn   = wave & 1;           // 0..1
  const int m0   = blockIdx.x * TBM;
  const int n0   = blockIdx.y * TBN;
  const bool kvec = ((K & 3) == 0);

  v8f acc0 = {}, acc1 = {};
  for (int kt = 0; kt < K; kt += TKT) {
    // ---- stage A tile 64x32 (512 v4h groups, 2 per thread) ----
    for (int t = tid; t < TBM * (TKT / 4); t += 256) {
      int g = t & 7, r = t >> 3;
      int gr = m0 + r, gc = kt + (g << 2);
      v4h h4 = {};
      if (gr < M) {
        if (kvec && gc + 3 < K) {
          float4 f4 = *(const float4*)(A + (size_t)gr * K + gc);
          h4[0] = (_Float16)f4.x; h4[1] = (_Float16)f4.y;
          h4[2] = (_Float16)f4.z; h4[3] = (_Float16)f4.w;
        } else {
#pragma unroll
          for (int j = 0; j < 4; ++j)
            if (gc + j < K) h4[j] = (_Float16)A[(size_t)gr * K + gc + j];
        }
      }
      *(v4h*)(sA + r * LDA + (g << 2)) = h4;
    }
    // ---- stage B tile 32x64 transposed into sBt[n][k] ----
    for (int t = tid; t < TBN * (TKT / 4); t += 256) {
      int c = t & 63, g = t >> 6;
      int kr = kt + (g << 2);
      v4h h4 = {};
#pragma unroll
      for (int j = 0; j < 4; ++j) {
        float v = (kr + j < K) ? W[(size_t)(kr + j) * Nc + n0 + c] : 0.f;
        h4[j] = (_Float16)v;
      }
      *(v4h*)(sBt + c * LDB + (g << 2)) = h4;
    }
    if (kt + TKT < K && tid < TKT)                     // global_prefetch_b8
      __builtin_prefetch(W + (size_t)(kt + TKT + tid) * Nc + n0, 0, 1);
    __syncthreads();
    // ---- 2 WMMAs per wave, shared A fragment ----
    const int kb = (lane >> 4) << 3;                   // 0 or 8
    v16h af = load_frag(sA + ((wm << 4) + (lane & 15)) * LDA, kb);
    v16h b0 = load_frag(sBt + ((wn << 5) + (lane & 15)) * LDB, kb);
    v16h b1 = load_frag(sBt + ((wn << 5) + 16 + (lane & 15)) * LDB, kb);
    acc0 = __builtin_amdgcn_wmma_f32_16x16x32_f16(false, af, false, b0,
                                                  (short)0, acc0, false, false);
    acc1 = __builtin_amdgcn_wmma_f32_16x16x32_f16(false, af, false, b1,
                                                  (short)0, acc1, false, false);
    __syncthreads();
  }
  // ---- epilogue: C VGPR r holds row r + 8*(lane>>4), col lane&15 ----
  const int cn0 = n0 + (wn << 5) + (lane & 15);
  const int cn1 = cn0 + 16;
  const float bv0 = bias ? bias[cn0] : 0.f;
  const float bv1 = bias ? bias[cn1] : 0.f;
  const int rbase = m0 + (wm << 4) + ((lane >> 4) << 3);
  if (rbase + 7 < M) {                                 // interior fast path
#pragma unroll
    for (int r = 0; r < 8; ++r) {
      size_t ro = (size_t)(rbase + r) * Nc;
      C[ro + cn0] = apply_act(acc0[r] + bv0, ACT);
      C[ro + cn1] = apply_act(acc1[r] + bv1, ACT);
    }
  } else {
#pragma unroll
    for (int r = 0; r < 8; ++r) {
      if (rbase + r < M) {
        size_t ro = (size_t)(rbase + r) * Nc;
        C[ro + cn0] = apply_act(acc0[r] + bv0, ACT);
        C[ro + cn1] = apply_act(acc1[r] + bv1, ACT);
      }
    }
  }
}

// ---------------------------------------------------------------------------
// Gate GEMM: m[E,256] = leaky( concat(x[src[e]], ea[e]) @ W1[320,256] )
// The concat boundary (256) is K-tile aligned: tiles 0..7 gather from x,
// tiles 8..9 from ea — both with aligned float4 reads.
// ---------------------------------------------------------------------------
__global__ __launch_bounds__(256)
void gemm_gate_wmma(const float* __restrict__ x, const float* __restrict__ ea,
                    const int* __restrict__ src, const float* __restrict__ W1,
                    float* __restrict__ C, int M)
{
  constexpr int KG = HD + 64;     // 320
  __shared__ __align__(16) _Float16 sA [TBM * LDA];
  __shared__ __align__(16) _Float16 sBt[TBN * LDB];
  const int tid  = threadIdx.x;
  const int lane = tid & 31;
  const int wave = tid >> 5;
  const int wm = wave >> 1, wn = wave & 1;
  const int m0 = blockIdx.x * TBM;
  const int n0 = blockIdx.y * TBN;

  v8f acc0 = {}, acc1 = {};
  for (int kt = 0; kt < KG; kt += TKT) {
    for (int t = tid; t < TBM * (TKT / 4); t += 256) {
      int g = t & 7, r = t >> 3;
      int e = m0 + r, gc = kt + (g << 2);
      v4h h4 = {};
      if (e < M) {
        float4 f4;
        if (kt < HD) f4 = *(const float4*)(x  + (size_t)src[e] * HD + gc);
        else         f4 = *(const float4*)(ea + (size_t)e * 64 + (gc - HD));
        h4[0] = (_Float16)f4.x; h4[1] = (_Float16)f4.y;
        h4[2] = (_Float16)f4.z; h4[3] = (_Float16)f4.w;
      }
      *(v4h*)(sA + r * LDA + (g << 2)) = h4;
    }
    for (int t = tid; t < TBN * (TKT / 4); t += 256) {
      int c = t & 63, g = t >> 6;
      int kr = kt + (g << 2);
      v4h h4 = {};
#pragma unroll
      for (int j = 0; j < 4; ++j)
        h4[j] = (_Float16)W1[(size_t)(kr + j) * HD + n0 + c];
      *(v4h*)(sBt + c * LDB + (g << 2)) = h4;
    }
    __syncthreads();
    const int kb = (lane >> 4) << 3;
    v16h af = load_frag(sA + ((wm << 4) + (lane & 15)) * LDA, kb);
    v16h b0 = load_frag(sBt + ((wn << 5) + (lane & 15)) * LDB, kb);
    v16h b1 = load_frag(sBt + ((wn << 5) + 16 + (lane & 15)) * LDB, kb);
    acc0 = __builtin_amdgcn_wmma_f32_16x16x32_f16(false, af, false, b0,
                                                  (short)0, acc0, false, false);
    acc1 = __builtin_amdgcn_wmma_f32_16x16x32_f16(false, af, false, b1,
                                                  (short)0, acc1, false, false);
    __syncthreads();
  }
  const int cn0 = n0 + (wn << 5) + (lane & 15);
  const int cn1 = cn0 + 16;
  const int rbase = m0 + (wm << 4) + ((lane >> 4) << 3);
  if (rbase + 7 < M) {
#pragma unroll
    for (int r = 0; r < 8; ++r) {
      size_t ro = (size_t)(rbase + r) * HD;
      C[ro + cn0] = leaky01(acc0[r]);
      C[ro + cn1] = leaky01(acc1[r]);
    }
  } else {
#pragma unroll
    for (int r = 0; r < 8; ++r) {
      if (rbase + r < M) {
        size_t ro = (size_t)(rbase + r) * HD;
        C[ro + cn0] = leaky01(acc0[r]);
        C[ro + cn1] = leaky01(acc1[r]);
      }
    }
  }
}

// ------------------------- irregular / elementwise kernels -----------------
__global__ void fill_f32(float* p, float v, long n) {
  long t = blockIdx.x * (long)blockDim.x + threadIdx.x;
  if (t < n) p[t] = v;
}

// out[row] = dot(A[row,:K], vec)  — one wave per row, shfl reduction
__global__ void rowdot(const float* __restrict__ A, const float* __restrict__ vec,
                       float* __restrict__ out, int rows, int K)
{
  int row  = blockIdx.x * 8 + (threadIdx.x >> 5);
  int lane = threadIdx.x & 31;
  if (row >= rows) return;
  float s = 0.f;
  for (int k = lane; k < K; k += 32) s += A[(size_t)row * K + k] * vec[k];
#pragma unroll
  for (int off = 16; off > 0; off >>= 1) s += __shfl_xor(s, off);
  if (lane == 0) out[row] = s;
}

__global__ void alpha_gate_max(const float* edot, const float* ndot, const int* dst,
                               float* alpha, float* smax, int n) {
  int e = blockIdx.x * blockDim.x + threadIdx.x;
  if (e >= n) return;
  float a = leaky01(edot[e] + ndot[dst[e]]);
  alpha[e] = a;
  atomicMaxF(&smax[dst[e]], a);
}

__global__ void alpha_conv_max(const float* asrcN, const float* adstN,
                               const int* src, const int* dst,
                               float* alpha, float* smax, int n) {
  int e = blockIdx.x * blockDim.x + threadIdx.x;
  if (e >= n) return;
  float a = leaky01(asrcN[src[e]] + adstN[dst[e]]);
  alpha[e] = a;
  atomicMaxF(&smax[dst[e]], a);
}

__global__ void alpha_mol_max(const float* a_s, const float* adot, const int* batch,
                              float* alpha, float* smax, int n) {
  int i = blockIdx.x * blockDim.x + threadIdx.x;
  if (i >= n) return;
  float a = leaky01(a_s[i] + adot[batch[i]]);
  alpha[i] = a;
  atomicMaxF(&smax[batch[i]], a);
}

__global__ void seg_exp_sum(float* alpha, const int* idx, const float* smax,
                            float* ssum, int n) {
  int i = blockIdx.x * blockDim.x + threadIdx.x;
  if (i >= n) return;
  float a = __expf(alpha[i] - smax[idx[i]]);
  alpha[i] = a;
  atomicAdd(&ssum[idx[i]], a);
}

__global__ void seg_norm(float* alpha, const int* idx, const float* ssum, int n) {
  int i = blockIdx.x * blockDim.x + threadIdx.x;
  if (i >= n) return;
  alpha[i] = alpha[i] / (ssum[idx[i]] + 1e-16f);
}

// acc[seg[item], f] += w[item] * rows[rowIdx?rowIdx[item]:item, f]   (F == 256)
__global__ void scatter_weighted(float* __restrict__ acc, const float* __restrict__ rows,
                                 const float* __restrict__ w, const int* __restrict__ seg,
                                 const int* __restrict__ rowIdx, long total)
{
  long t = blockIdx.x * (long)blockDim.x + threadIdx.x;
  if (t >= total) return;
  int item = (int)(t >> 8);
  int f    = (int)(t & 255);
  int r    = rowIdx ? rowIdx[item] : item;
  float wv = w ? w[item] : 1.f;
  atomicAdd(&acc[(size_t)seg[item] * HD + f], wv * rows[(size_t)r * HD + f]);
}

__global__ void bias_elu_k(float* a, const float* __restrict__ bias, long total) {
  long t = blockIdx.x * (long)blockDim.x + threadIdx.x;
  if (t >= total) return;
  a[t] = elu1(a[t] + bias[t & 255]);
}

__global__ void relu_k(float* a, long n) {
  long t = blockIdx.x * (long)blockDim.x + threadIdx.x;
  if (t < n) a[t] = fmaxf(a[t], 0.f);
}

// x_out = relu(GRUCell(input=h -> gi, hidden=x -> gh))
__global__ void gru_relu_k(const float* __restrict__ gi, const float* __restrict__ gh,
                           const float* __restrict__ hid, float* __restrict__ outx,
                           int rows)
{
  long t = blockIdx.x * (long)blockDim.x + threadIdx.x;
  long total = (long)rows * HD;
  if (t >= total) return;
  int row = (int)(t >> 8), f = (int)(t & 255);
  const float* giR = gi + (size_t)row * GH3;
  const float* ghR = gh + (size_t)row * GH3;
  float r  = sigmf(giR[f] + ghR[f]);
  float z  = sigmf(giR[HD + f] + ghR[HD + f]);
  float nn = tanhf(giR[2 * HD + f] + r * ghR[2 * HD + f]);
  float v  = (1.f - z) * nn + z * hid[t];
  outx[t] = fmaxf(v, 0.f);
}

// out[row] = dot(A[row,:K], w) + b[0]  (K <= 128, head layers)
__global__ void gemv_head(const float* __restrict__ A, const float* __restrict__ w,
                          const float* __restrict__ b, float* __restrict__ out,
                          int rows, int K)
{
  int i = blockIdx.x * blockDim.x + threadIdx.x;
  if (i >= rows) return;
  float s = b[0];
  for (int k = 0; k < K; ++k) s += A[(size_t)i * K + k] * w[k];
  out[i] = s;
}

// ---------------------------------------------------------------------------
extern "C" void kernel_launch(void* const* d_in, const int* in_sizes, int n_in,
                              void* d_out, int out_size, void* d_ws, size_t ws_size,
                              hipStream_t stream)
{
  (void)n_in; (void)ws_size;
  const int N = in_sizes[3];          // batch vector length
  const int E = in_sizes[2] / 2;      // edge_index [2,E]
  const int B = out_size - N;         // sens[B] ++ mie[N]
  const int H = HD;

  const float* x_raw = (const float*)d_in[0];
  const float* ebond = (const float*)d_in[1];
  const int*   eidx  = (const int*)d_in[2];
  const int*   src   = eidx;
  const int*   dst   = eidx + E;
  const int*   batch = (const int*)d_in[3];

  int p = 4;
  auto nf = [&]() { return (const float*)d_in[p++]; };
  const float *aeW = nf(), *aeB = nf();                 // atom_enc
  const float *beW = nf(), *beB = nf();                 // bond_enc
  const float *l1W = nf(), *l1B = nf();                 // lin1
  const float *gW1 = nf(), *gW2 = nf(), *gAl = nf(), *gAr = nf(), *gBias = nf();
  const float *g0Wi = nf(), *g0Wh = nf(), *g0bi = nf(), *g0bh = nf();
  const float *cW[2], *cAs[2], *cAd[2], *cB_[2];
  for (int c = 0; c < 2; ++c) { cW[c]=nf(); cAs[c]=nf(); cAd[c]=nf(); cB_[c]=nf(); }
  const float *gWi[2], *gWh[2], *gbi[2], *gbh[2];
  for (int c = 0; c < 2; ++c) { gWi[c]=nf(); gWh[c]=nf(); gbi[c]=nf(); gbh[c]=nf(); }
  const float *mcW = nf(), *mcAs = nf(), *mcAd = nf(), *mcB = nf();
  const float *mgWi = nf(), *mgWh = nf(), *mgbi = nf(), *mgbh = nf();
  const float *l2W = nf(), *l2B = nf();
  const float *pjW = nf(), *pjB = nf();
  const float *s1W = nf(), *s1B = nf();
  const float *s2W = nf(), *s2B = nf();
  const float *m1W = nf(), *m1B = nf();
  const float *m2W = nf(), *m2B = nf();

  // ---- workspace layout (floats); big regions are time-shared -------------
  float* W = (float*)d_ws;
  size_t o = 0;
  auto alloc = [&](size_t nfl) { size_t r = o; o += nfl; return W + r; };
  float* f_x    = alloc((size_t)N * H);                        // node state
  float* f_h    = alloc((size_t)N * H);                        // h / tmp
  float* f_acc  = alloc((size_t)N * H);                        // seg accumulator
  float* f_hx   = alloc((size_t)N * H);                        // hx / hs
  size_t big = (size_t)N * GH3 > (size_t)E * H ? (size_t)N * GH3 : (size_t)E * H;
  float* f_big1 = alloc(big);                                  // m  -> gi
  float* f_big2 = alloc((size_t)N * GH3);                      // ea -> gh
  float* f_ea   = f_big2;                                      // E*64 <= N*3H
  float* f_alpha= alloc((size_t)E);
  float* f_edot = alloc((size_t)E);
  float* f_nd1  = alloc((size_t)N);
  float* f_nd2  = alloc((size_t)N);
  float* f_smax = alloc((size_t)N);
  float* f_ssum = alloc((size_t)N);
  float* f_adot = alloc((size_t)B);
  float* f_out  = alloc((size_t)B * H);
  float* f_molh = alloc((size_t)B * H);
  float* f_gim  = alloc((size_t)B * GH3);
  float* f_ghm  = alloc((size_t)B * GH3);
  float* f_ge   = alloc((size_t)B * H);
  float* f_ge2  = alloc((size_t)B * 64);
  float* f_s1   = alloc((size_t)B * 128);
  float* f_m1   = alloc((size_t)N * 64);

  auto g1 = [](long n) { return dim3((unsigned)((n + 255) / 256)); };
  auto gemm = [&](const float* A, const float* Wm, const float* bias, float* C,
                  int M, int K, int Nc, int act) {
    dim3 g((M + TBM - 1) / TBM, Nc / TBN);
    switch (act) {
      case 0: gemm_wmma<0><<<g, 256, 0, stream>>>(A, Wm, bias, C, M, K, Nc); break;
      case 1: gemm_wmma<1><<<g, 256, 0, stream>>>(A, Wm, bias, C, M, K, Nc); break;
      case 2: gemm_wmma<2><<<g, 256, 0, stream>>>(A, Wm, bias, C, M, K, Nc); break;
      default: gemm_wmma<3><<<g, 256, 0, stream>>>(A, Wm, bias, C, M, K, Nc); break;
    }
  };

  // ====================== encoders + lin1 ==================================
  gemm(x_raw, aeW, aeB, f_h, N, 39, H, 0);                     // atom encoder
  gemm(f_h, l1W, l1B, f_x, N, H, H, 2);                        // lin1 + leaky
  gemm(ebond, beW, beB, f_ea, E, 10, 64, 0);                   // bond encoder

  // ====================== GATEConv =========================================
  { dim3 g((E + TBM - 1) / TBM, H / TBN);
    gemm_gate_wmma<<<g, 256, 0, stream>>>(f_x, f_ea, src, gW1, f_big1, E); }
  rowdot<<<dim3((E + 7) / 8), 256, 0, stream>>>(f_big1, gAl, f_edot, E, H);
  rowdot<<<dim3((N + 7) / 8), 256, 0, stream>>>(f_x, gAr, f_nd1, N, H);
  fill_f32<<<g1(N), 256, 0, stream>>>(f_smax, -1e30f, N);
  fill_f32<<<g1(N), 256, 0, stream>>>(f_ssum, 0.f, N);
  alpha_gate_max<<<g1(E), 256, 0, stream>>>(f_edot, f_nd1, dst, f_alpha, f_smax, E);
  seg_exp_sum<<<g1(E), 256, 0, stream>>>(f_alpha, dst, f_smax, f_ssum, E);
  seg_norm<<<g1(E), 256, 0, stream>>>(f_alpha, dst, f_ssum, E);
  fill_f32<<<g1((long)N * H), 256, 0, stream>>>(f_acc, 0.f, (long)N * H);
  scatter_weighted<<<g1((long)E * H), 256, 0, stream>>>(f_acc, f_big1, f_alpha, dst,
                                                        nullptr, (long)E * H);
  gemm(f_acc, gW2, gBias, f_h, N, H, H, 3);                    // (Σαm)@W2+b, elu
  gemm(f_h, g0Wi, g0bi, f_big1, N, H, GH3, 0);                 // gi  (m is dead)
  gemm(f_x, g0Wh, g0bh, f_big2, N, H, GH3, 0);                 // gh  (ea is dead)
  gru_relu_k<<<g1((long)N * H), 256, 0, stream>>>(f_big1, f_big2, f_x, f_x, N);

  // ====================== 2x GATConv + GRU =================================
  for (int c = 0; c < 2; ++c) {
    gemm(f_x, cW[c], nullptr, f_hx, N, H, H, 0);               // hx = x@W
    rowdot<<<dim3((N + 7) / 8), 256, 0, stream>>>(f_hx, cAs[c], f_nd1, N, H);
    rowdot<<<dim3((N + 7) / 8), 256, 0, stream>>>(f_hx, cAd[c], f_nd2, N, H);
    fill_f32<<<g1(N), 256, 0, stream>>>(f_smax, -1e30f, N);
    fill_f32<<<g1(N), 256, 0, stream>>>(f_ssum, 0.f, N);
    alpha_conv_max<<<g1(E), 256, 0, stream>>>(f_nd1, f_nd2, src, dst, f_alpha, f_smax, E);
    seg_exp_sum<<<g1(E), 256, 0, stream>>>(f_alpha, dst, f_smax, f_ssum, E);
    seg_norm<<<g1(E), 256, 0, stream>>>(f_alpha, dst, f_ssum, E);
    fill_f32<<<g1((long)N * H), 256, 0, stream>>>(f_acc, 0.f, (long)N * H);
    scatter_weighted<<<g1((long)E * H), 256, 0, stream>>>(f_acc, f_hx, f_alpha, dst,
                                                          src, (long)E * H);
    bias_elu_k<<<g1((long)N * H), 256, 0, stream>>>(f_acc, cB_[c], (long)N * H);
    gemm(f_acc, gWi[c], gbi[c], f_big1, N, H, GH3, 0);
    gemm(f_x, gWh[c], gbh[c], f_big2, N, H, GH3, 0);
    gru_relu_k<<<g1((long)N * H), 256, 0, stream>>>(f_big1, f_big2, f_x, f_x, N);
  }

  // ====================== molecule (graph) pass ============================
  fill_f32<<<g1((long)B * H), 256, 0, stream>>>(f_out, 0.f, (long)B * H);
  scatter_weighted<<<g1((long)N * H), 256, 0, stream>>>(f_out, f_x, nullptr, batch,
                                                        nullptr, (long)N * H);
  relu_k<<<g1((long)B * H), 256, 0, stream>>>(f_out, (long)B * H);
  gemm(f_x, mcW, nullptr, f_hx, N, H, H, 0);                   // hs
  rowdot<<<dim3((N + 7) / 8), 256, 0, stream>>>(f_hx, mcAs, f_nd1, N, H);  // a_s
  for (int t = 0; t < 2; ++t) {
    gemm(f_out, mcW, nullptr, f_molh, B, H, H, 0);             // hd
    rowdot<<<dim3((B + 7) / 8), 256, 0, stream>>>(f_molh, mcAd, f_adot, B, H);
    fill_f32<<<g1(B), 256, 0, stream>>>(f_smax, -1e30f, B);
    fill_f32<<<g1(B), 256, 0, stream>>>(f_ssum, 0.f, B);
    alpha_mol_max<<<g1(N), 256, 0, stream>>>(f_nd1, f_adot, batch, f_alpha, f_smax, N);
    seg_exp_sum<<<g1(N), 256, 0, stream>>>(f_alpha, batch, f_smax, f_ssum, N);
    seg_norm<<<g1(N), 256, 0, stream>>>(f_alpha, batch, f_ssum, N);
    fill_f32<<<g1((long)B * H), 256, 0, stream>>>(f_molh, 0.f, (long)B * H);
    scatter_weighted<<<g1((long)N * H), 256, 0, stream>>>(f_molh, f_hx, f_alpha, batch,
                                                          nullptr, (long)N * H);
    bias_elu_k<<<g1((long)B * H), 256, 0, stream>>>(f_molh, mcB, (long)B * H);
    gemm(f_molh, mgWi, mgbi, f_gim, B, H, GH3, 0);
    gemm(f_out, mgWh, mgbh, f_ghm, B, H, GH3, 0);
    gru_relu_k<<<g1((long)B * H), 256, 0, stream>>>(f_gim, f_ghm, f_out, f_out, B);
  }

  // ====================== heads ============================================
  gemm(f_out, l2W, l2B, f_ge, B, H, H, 0);                     // lin2
  gemm(f_ge, pjW, pjB, f_ge2, B, H, 64, 1);                    // proj + relu
  gemm(f_ge2, s1W, s1B, f_s1, B, 64, 128, 1);                  // sens1 + relu
  gemv_head<<<g1(B), 256, 0, stream>>>(f_s1, s2W, s2B, (float*)d_out, B, 128);
  gemm(f_x, m1W, m1B, f_m1, N, H, 64, 1);                      // mie1 + relu
  gemv_head<<<g1(N), 256, 0, stream>>>(f_m1, m2W, m2B, (float*)d_out + B, N, 64);
}
